// PPRGoEmmbeddingDiffusions_59296318488772
// MI455X (gfx1250) — matrix operand order repaired and checked by reading, not compiled, not run on hardware
//
#include <hip/hip_runtime.h>
#include <hip/hip_bf16.h>

// ---------------------------------------------------------------------------
// PPRGo embedding diffusion on MI455X (gfx1250, wave32).
//
//   P1[b]   = sum_{r in seg b} s_r * relu(X[r] @ W1)   (fused GEMM + reduce)
//   prop    = P1 @ W2          (uses segsum(relu(XW1)W2*s) == segsum(relu(XW1)*s)@W2,
//   T       = relu(prop @ W3)   collapsing the 1M-row W2 GEMM to 8192 rows)
//   logits  = T @ W4
//
// Roofline: stage 1 = 137 GFLOP vs 1 GB of X @ 23.3 TB/s (43 us). fp32 WMMA
// (16x16x4) is ~5x too slow; bf16 WMMA (16x16x32) is 8x denser. We use bf16x3
// error-compensated splitting (Ah*Bh + Al*Bh + Ah*Bl, f32 accum) for ~fp32
// fidelity (~16-17 mantissa bits) at ~3x bf16 cost -> roughly HBM-balanced.
// Conversions happen once into LDS bf16 planes, amortized across all waves.
// Small GEMMs (2.4 GFLOP total) stay fp32 WMMA and stage via
// global_load_async_to_lds_b128 (ASYNCcnt) -- the CDNA5 async path.
// ---------------------------------------------------------------------------

#define NUM_PPR 1048576
#define BATCH   8192
#define TOPK    128
#define KDIM    256
#define NCLS    64

typedef __attribute__((ext_vector_type(2)))  float  v2f;
typedef __attribute__((ext_vector_type(8)))  float  v8f;
typedef __attribute__((ext_vector_type(8)))  __bf16 v8bf;
typedef __attribute__((ext_vector_type(16))) __bf16 v16bf;

__device__ __forceinline__ v8f wmma_f32_16x16x4(v2f a, v2f b, v8f c) {
  return __builtin_amdgcn_wmma_f32_16x16x4_f32(false, a, false, b, (short)0, c,
                                               false, false);
}
__device__ __forceinline__ v8f wmma_bf16(v16bf a, v16bf b, v8f c) {
  return __builtin_amdgcn_wmma_f32_16x16x32_bf16(false, a, false, b, (short)0,
                                                 c, false, false);
}

// --- CDNA5 async global->LDS copies (ASYNCcnt) -----------------------------
__device__ __forceinline__ unsigned lds_off(const void* p) {
  // generic LDS addr: low 32 bits are the wave-relative LDS byte address
  return (unsigned)(unsigned long long)p;
}
__device__ __forceinline__ void async_ld_b128(unsigned ldsaddr, const void* g) {
  asm volatile("global_load_async_to_lds_b128 %0, %1, off"
               :: "v"(ldsaddr), "v"(g) : "memory");
}
__device__ __forceinline__ void async_ld_b32(unsigned ldsaddr, const void* g) {
  asm volatile("global_load_async_to_lds_b32 %0, %1, off"
               :: "v"(ldsaddr), "v"(g) : "memory");
}
__device__ __forceinline__ void wait_async0() {
  asm volatile("s_wait_asynccnt 0x0" ::: "memory");
}

// --- fp32 -> (hi, lo) bf16 split, packed as u32 (low 16 bits = lower K) ----
__device__ __forceinline__ unsigned pack2(__bf16 a, __bf16 b) {
  union { __bf16 h[2]; unsigned u; } p;
  p.h[0] = a; p.h[1] = b;
  return p.u;
}
__device__ __forceinline__ void split_bf16(float x, __bf16& hi, __bf16& lo) {
  hi = (__bf16)x;
  lo = (__bf16)(x - (float)hi);
}

// ---------------------------------------------------------------------------
// Kernel 1: P1[b,:] = sum_r s_r * relu(X[128 rows of seg b] @ W1)  (bf16x3)
// grid = BATCH blocks, 256 threads (8 waves: 2(M) x 4(N)).
// LDS: Xh,Xl planes [128][128] u32 (k-pairs)     2 * 64 KB
//      W panels hi/lo [2 buf][16][256] u32        2 * 32 KB   (K=32 panels)
//      scores[128], red[2][256]                   ~2.5 KB     => ~195 KB
// ---------------------------------------------------------------------------
__global__ __launch_bounds__(256) void ppr_stage1_kernel(
    const float* __restrict__ X, const float* __restrict__ scores,
    const float* __restrict__ W1, float* __restrict__ P1) {
  extern __shared__ unsigned smem_u[];
  unsigned* xh  = smem_u;                 // 128*128 words
  unsigned* xl  = xh + 128 * 128;         // 128*128 words
  unsigned* whi = xl + 128 * 128;         // 2 * 16*256 words
  unsigned* wlo = whi + 2 * 16 * 256;     // 2 * 16*256 words
  float* lds_s  = (float*)(wlo + 2 * 16 * 256);  // 128
  float* lds_r  = lds_s + 128;                   // 512

  const int tid  = threadIdx.x;
  const int lane = tid & 31;
  const int wave = tid >> 5;
  const int wm   = wave >> 2;             // 0..1 : 64-row M half
  const int wn   = wave & 3;              // 0..3 : 64-col N quarter
  const int half = lane >> 4;
  const int l16  = lane & 15;

  const size_t row0 = (size_t)blockIdx.x * TOPK;

  // ---- ppr scores: async b32 global->LDS ----------------------------------
  if (tid < TOPK) async_ld_b32(lds_off(lds_s + tid), scores + row0 + tid);

  // ---- stage X tile as bf16 hi/lo planes (load -> split -> ds_store) ------
  {
    const int kp = tid & 127;             // k-pair (2 floats)
    const int m0 = tid >> 7;              // 0..1
#pragma unroll 4
    for (int i = 0; i < 64; ++i) {
      const int m = 2 * i + m0;
      const float2 xv =
          *(const float2*)&X[(row0 + m) * KDIM + 2 * kp];
      __bf16 h0, l0, h1, l1;
      split_bf16(xv.x, h0, l0);
      split_bf16(xv.y, h1, l1);
      xh[m * 128 + kp] = pack2(h0, h1);
      xl[m * 128 + kp] = pack2(l0, l1);
    }
  }

  // ---- W1 panel 0 -> regs -> bf16 planes (buffer 0) -----------------------
  float wreg[32];
#pragma unroll
  for (int j = 0; j < 32; ++j) wreg[j] = W1[(size_t)j * KDIM + tid];
#pragma unroll
  for (int j = 0; j < 16; ++j) {
    __bf16 h0, l0, h1, l1;
    split_bf16(wreg[2 * j], h0, l0);
    split_bf16(wreg[2 * j + 1], h1, l1);
    whi[j * 256 + tid] = pack2(h0, h1);
    wlo[j * 256 + tid] = pack2(l0, l1);
  }

  const v8f vzero = {};
  v8f acc[4][4];
  for (int mt = 0; mt < 4; ++mt)
    for (int nt = 0; nt < 4; ++nt) acc[mt][nt] = vzero;

  wait_async0();      // scores landed
  __syncthreads();    // X planes + panel 0 visible

  // ---- K loop: 8 panels of K=32, double-buffered W planes -----------------
  for (int p = 0; p < 8; ++p) {
    const int buf = p & 1;
    if (p < 7) {  // next panel's globals in flight during compute
#pragma unroll
      for (int j = 0; j < 32; ++j)
        wreg[j] = W1[(size_t)((p + 1) * 32 + j) * KDIM + tid];
      if (p < 6)
        __builtin_prefetch(W1 + (size_t)(p + 2) * 32 * KDIM + tid, 0, 1);
    }

    const int kw0 = p * 16;  // word (k-pair) base into X planes
    v16bf Ah[4], Al[4];
#pragma unroll
    for (int mt = 0; mt < 4; ++mt) {
      const int m = wm * 64 + mt * 16 + l16;
      const int idx = m * 128 + kw0 + 4 * half;
      v8bf h0 = *(const v8bf*)(xh + idx);
      v8bf h1 = *(const v8bf*)(xh + idx + 8);
      v8bf l0 = *(const v8bf*)(xl + idx);
      v8bf l1 = *(const v8bf*)(xl + idx + 8);
      Ah[mt] = __builtin_shufflevector(h0, h1, 0, 1, 2, 3, 4, 5, 6, 7, 8, 9,
                                       10, 11, 12, 13, 14, 15);
      Al[mt] = __builtin_shufflevector(l0, l1, 0, 1, 2, 3, 4, 5, 6, 7, 8, 9,
                                       10, 11, 12, 13, 14, 15);
    }
#pragma unroll
    for (int nt = 0; nt < 4; ++nt) {
      const int n = wn * 64 + nt * 16 + l16;
      union { unsigned u[8]; v16bf v; } bh, bl;
#pragma unroll
      for (int j = 0; j < 8; ++j) {
        const int off = buf * 4096 + (4 * half + j + (j & 4)) * 256 + n;
        bh.u[j] = whi[off];
        bl.u[j] = wlo[off];
      }
#pragma unroll
      for (int mt = 0; mt < 4; ++mt) {
        acc[mt][nt] = wmma_bf16(Ah[mt], bh.v, acc[mt][nt]);  // hi*hi
        acc[mt][nt] = wmma_bf16(Al[mt], bh.v, acc[mt][nt]);  // lo*hi
        acc[mt][nt] = wmma_bf16(Ah[mt], bl.v, acc[mt][nt]);  // hi*lo
      }
    }

    if (p < 7) {  // convert + store next panel into the other buffer
      const int nb = (p + 1) & 1;
#pragma unroll
      for (int j = 0; j < 16; ++j) {
        __bf16 h0, l0, h1, l1;
        split_bf16(wreg[2 * j], h0, l0);
        split_bf16(wreg[2 * j + 1], h1, l1);
        whi[nb * 4096 + j * 256 + tid] = pack2(h0, h1);
        wlo[nb * 4096 + j * 256 + tid] = pack2(l0, l1);
      }
    }
    __syncthreads();
  }

  // ---- epilogue: relu, ppr weight, reduce over all 128 rows ---------------
  float sc[4][8];
#pragma unroll
  for (int mt = 0; mt < 4; ++mt)
#pragma unroll
    for (int j = 0; j < 8; ++j)
      sc[mt][j] = lds_s[wm * 64 + mt * 16 + half * 8 + j];

#pragma unroll
  for (int nt = 0; nt < 4; ++nt) {
    float part = 0.f;
#pragma unroll
    for (int mt = 0; mt < 4; ++mt) {
      const float* ap = (const float*)&acc[mt][nt];
#pragma unroll
      for (int j = 0; j < 8; ++j) part += fmaxf(ap[j], 0.f) * sc[mt][j];
    }
    part += __shfl_xor(part, 16, 32);
    if (lane < 16) lds_r[wm * 256 + wn * 64 + nt * 16 + l16] = part;
  }
  __syncthreads();
  P1[(size_t)blockIdx.x * KDIM + tid] = lds_r[tid] + lds_r[256 + tid];
}

// ---------------------------------------------------------------------------
// fp32 K=256 GEMM: C[M,BN] = act(A[M,256] @ B[256,BN]), 128 rows/block.
// A tile + B panels staged with global_load_async_to_lds_b128 (ASYNCcnt).
// ---------------------------------------------------------------------------
template <int BN, bool RELU>
__global__ __launch_bounds__(256) void gemm_k256_kernel(
    const float* __restrict__ A, const float* __restrict__ B,
    float* __restrict__ C) {
  extern __shared__ float smem[];
  float* lds_a = smem;
  float* lds_b = lds_a + 128 * KDIM;

  constexpr int WN = BN / 4;
  constexpr int NT = WN / 16;

  const int tid  = threadIdx.x;
  const int lane = tid & 31;
  const int wave = tid >> 5;
  const int wm   = wave >> 2;
  const int wn   = wave & 3;
  const int half = lane >> 4;
  const int l16  = lane & 15;

  const size_t row0 = (size_t)blockIdx.x * 128;

  {  // async-stage A tile (128x256 f32)
    const float4* ga = (const float4*)(A + row0 * KDIM);
    const unsigned la = lds_off(lds_a);
    for (int i = tid; i < 128 * (KDIM / 4); i += 256)
      async_ld_b128(la + i * 16, ga + i);
  }

  const v8f vzero = {};
  v8f acc[4][NT];
  for (int mt = 0; mt < 4; ++mt)
    for (int nt = 0; nt < NT; ++nt) acc[mt][nt] = vzero;

  const unsigned lb = lds_off(lds_b);
  for (int p = 0; p < KDIM / 16; ++p) {
    const int k0 = p * 16;
    __syncthreads();  // previous panel consumed
    {
      const float4* gb = (const float4*)(B + (size_t)k0 * BN);
      for (int i = tid; i < 16 * (BN / 4); i += 256)
        async_ld_b128(lb + i * 16, gb + i);
      if (p + 1 < KDIM / 16)
        __builtin_prefetch(B + (size_t)(k0 + 16) * BN, 0, 1);
    }
    wait_async0();    // panel (and, first iter, A tile) landed
    __syncthreads();

    for (int kk = 0; kk < 16; kk += 4) {
      const int kb = kk + (half << 1);
      v2f afrag[4];
#pragma unroll
      for (int mt = 0; mt < 4; ++mt) {
        const int m = wm * 64 + mt * 16 + l16;
        afrag[mt] = *(const v2f*)&lds_a[m * KDIM + k0 + kb];
      }
#pragma unroll
      for (int nt = 0; nt < NT; ++nt) {
        const int n = wn * WN + nt * 16 + l16;
        v2f bfrag;
        bfrag.x = lds_b[kb * BN + n];
        bfrag.y = lds_b[(kb + 1) * BN + n];
#pragma unroll
        for (int mt = 0; mt < 4; ++mt)
          acc[mt][nt] = wmma_f32_16x16x4(afrag[mt], bfrag, acc[mt][nt]);
      }
    }
  }

#pragma unroll
  for (int mt = 0; mt < 4; ++mt) {
    const size_t mbase = row0 + wm * 64 + mt * 16 + half * 8;
#pragma unroll
    for (int nt = 0; nt < NT; ++nt) {
      const int n = wn * WN + nt * 16 + l16;
      const float* ap = (const float*)&acc[mt][nt];
#pragma unroll
      for (int j = 0; j < 8; ++j) {
        float v = ap[j];
        if (RELU) v = fmaxf(v, 0.f);
        C[(mbase + j) * BN + n] = v;
      }
    }
  }
}

// ---------------------------------------------------------------------------
// Host launcher. Inputs: 0:X 1:ppr_scores 2:ppr_idx (unused; sorted fixed-128
// segments) 3:W1 4:W2 5:W3 6:W4.  Output: logits [8192,64] f32.
// ---------------------------------------------------------------------------
extern "C" void kernel_launch(void* const* d_in, const int* in_sizes, int n_in,
                              void* d_out, int out_size, void* d_ws,
                              size_t ws_size, hipStream_t stream) {
  const float* X  = (const float*)d_in[0];
  const float* s  = (const float*)d_in[1];
  const float* W1 = (const float*)d_in[3];
  const float* W2 = (const float*)d_in[4];
  const float* W3 = (const float*)d_in[5];
  const float* W4 = (const float*)d_in[6];
  float* out = (float*)d_out;

  float* buf0 = (float*)d_ws;                 // P1 / T
  float* buf1 = buf0 + (size_t)BATCH * KDIM;  // prop

  const size_t smem1 =
      (2 * 128 * 128 + 2 * 2 * 16 * 256 + 128 + 512) * sizeof(unsigned);
  const size_t smemA = (128 * KDIM + 16 * 256) * sizeof(float);
  const size_t smemB = (128 * KDIM + 16 * 64) * sizeof(float);

  ppr_stage1_kernel<<<BATCH, 256, smem1, stream>>>(X, s, W1, buf0);
  gemm_k256_kernel<256, false><<<BATCH / 128, 256, smemA, stream>>>(buf0, W2, buf1);
  gemm_k256_kernel<256, true><<<BATCH / 128, 256, smemA, stream>>>(buf1, W3, buf0);
  gemm_k256_kernel<64, false><<<BATCH / 128, 256, smemB, stream>>>(buf0, W4, out);
}
